// CustomAttention_33809982554433
// MI455X (gfx1250) — matrix operand order repaired and checked by reading, not compiled, not run on hardware
//
#include <hip/hip_runtime.h>
#include <hip/hip_bf16.h>

typedef __bf16 bf16_t;
typedef bf16_t v16bf __attribute__((ext_vector_type(16)));
typedef bf16_t v8bf  __attribute__((ext_vector_type(8)));
typedef float  v8f   __attribute__((ext_vector_type(8)));

#define D_MODEL 1024
#define NUM_HEAD 16
#define HID 64
#define SEQ 2048
#define BS 2
#define MROWS (SEQ * BS)      // 4096
#define BH (BS * NUM_HEAD)    // 32
#define INV_T 0.125f          // 1/sqrt(64)

#define KCH 64                // keys per staged LDS chunk
#define KP  72                // padded LDS row stride (elems): 144B rows, 16B aligned
#define NCHUNK (SEQ / KCH)    // 32

// ---------------- WMMA helper ----------------
static __device__ __forceinline__ v8f wmma_bf16(v16bf a, v16bf b, v8f c) {
  return __builtin_amdgcn_wmma_f32_16x16x32_bf16(false, a, false, b, (short)0, c,
                                                 false, false);
}

// A fragment 16x32 (M x K), source f32 row-major (ld elements).
// ISA 7.12.2: lane 0-15 -> M=lane, K={0..7,16..23}; lane 16-31 -> M=lane-16,
// K={8..15,24..31}.  element j -> k = j + (j & 8) + 8*sel.
static __device__ __forceinline__ v16bf load_a_f32(const float* __restrict__ src,
                                                   long ld, long m0, long k0, int lane) {
  long m = m0 + (lane & 15);
  const float* p = src + m * ld + k0 + ((lane >> 4) << 3);
  v16bf a;
#pragma unroll
  for (int j = 0; j < 16; ++j) a[j] = (bf16_t)p[j + (j & 8)];
  return a;
}

// Same fragment from bf16 row-major memory (two 16B chunks per lane).
static __device__ __forceinline__ v16bf load_a_bf16(const bf16_t* src, long ld,
                                                    long m0, long k0, int lane) {
  long m = m0 + (lane & 15);
  const bf16_t* p = src + m * ld + k0 + ((lane >> 4) << 3);
  v8bf lo = *(const v8bf*)(p);
  v8bf hi = *(const v8bf*)(p + 16);
  v16bf a;
#pragma unroll
  for (int j = 0; j < 8; ++j) { a[j] = lo[j]; a[j + 8] = hi[j]; }
  return a;
}

// B fragment 32x16 (K x N), source bf16 [k][n] row-major (ldn elements).
// Layout: N = lane & 15; element j -> k = j + (j & 8) + 8*sel (same as A).
static __device__ __forceinline__ v16bf load_b_bf16(const bf16_t* src, long ldn,
                                                    long k0, long n0, int lane) {
  const bf16_t* p = src + (k0 + ((lane >> 4) << 3)) * ldn + n0 + (lane & 15);
  v16bf b;
#pragma unroll
  for (int j = 0; j < 16; ++j) b[j] = p[(long)(j + (j & 8)) * ldn];
  return b;
}

// ---------------- CDNA5 async global->LDS copy (16B per lane) ----------------
static __device__ __forceinline__ void async_copy16(const bf16_t* g, bf16_t* l) {
  unsigned lofs = (unsigned)(size_t)l;                 // LDS aperture: addr[31:0]
  unsigned long long ga = (unsigned long long)(size_t)g;
  asm volatile("global_load_async_to_lds_b128 %0, %1, off"
               :: "v"(lofs), "v"(ga) : "memory");
}
#define WAIT_ASYNC0() asm volatile("s_wait_asynccnt 0" ::: "memory")
#define WAIT_ASYNC4() asm volatile("s_wait_asynccnt 4" ::: "memory")
#define WAIT_ASYNC8() asm volatile("s_wait_asynccnt 8" ::: "memory")

// Stage one 64-key chunk of Kt head (64 d x 64 keys bf16) into kb[d][key] (+pad).
// 128 threads x 4 async b128 each.
static __device__ __forceinline__ void stage_k(const bf16_t* Kh, int k0, bf16_t* kb,
                                               int t) {
#pragma unroll
  for (int i = 0; i < 4; ++i) {
    int c = i * 128 + t;
    int d = c >> 3, seg = c & 7;
    async_copy16(Kh + (long)d * SEQ + k0 + seg * 8, kb + d * KP + seg * 8);
  }
}

// Stage one 64-key chunk of V head (64 keys x 64 d bf16) into vb[key][d] (+pad).
static __device__ __forceinline__ void stage_v(const bf16_t* comb, long b, long h,
                                               int k0, bf16_t* vb, int t) {
#pragma unroll
  for (int i = 0; i < 4; ++i) {
    int c = i * 128 + t;
    int key = c >> 3, seg = c & 7;
    async_copy16(comb + ((long)(k0 + key) * BS + b) * (2 * D_MODEL) + D_MODEL +
                     h * HID + seg * 8,
                 vb + key * KP + seg * 8);
  }
}

// ---------------- kernel 0: weight transpose + bf16 cast:  W[n][k] -> Wt[k][n]
__global__ void wt_bf16_kernel(const float* __restrict__ W, bf16_t* __restrict__ Wt,
                               int N, int K) {
  long idx = (long)blockIdx.x * blockDim.x + threadIdx.x;
  if (idx >= (long)N * K) return;
  long k = idx / N, n = idx - k * N;
  Wt[idx] = (bf16_t)W[n * K + k];
}

// ---------------- kernel 0b: cast V into right half of combined [4096 x 2048]
__global__ void vcast_kernel(const float* __restrict__ V, bf16_t* __restrict__ comb) {
  long idx = (long)blockIdx.x * blockDim.x + threadIdx.x;
  if (idx >= (long)MROWS * D_MODEL) return;
  long m = idx >> 10, c = idx & 1023;
  comb[m * (2 * D_MODEL) + D_MODEL + c] = (bf16_t)V[idx];
}

// ---------------- kernel 1: Qp/Kp projection GEMM (M=4096,N=1024,K=1024)
__global__ void proj_qk_kernel(const float* __restrict__ X, const bf16_t* __restrict__ Wt,
                               const float* __restrict__ bias, bf16_t* __restrict__ outQ,
                               bf16_t* __restrict__ outKt, int isK) {
  int lane = threadIdx.x & 31;
  int wid  = threadIdx.x >> 5;
  long m0 = (long)blockIdx.x * 64 + (wid >> 1) * 32;
  long n0 = (long)blockIdx.y * 64 + (wid & 1) * 32;
  v8f acc[2][2] = {};
  for (int k = 0; k < D_MODEL; k += 32) {
    v16bf a0 = load_a_f32(X, D_MODEL, m0, k, lane);
    v16bf a1 = load_a_f32(X, D_MODEL, m0 + 16, k, lane);
    v16bf b0 = load_b_bf16(Wt, D_MODEL, k, n0, lane);
    v16bf b1 = load_b_bf16(Wt, D_MODEL, k, n0 + 16, lane);
    acc[0][0] = wmma_bf16(a0, b0, acc[0][0]);
    acc[0][1] = wmma_bf16(a0, b1, acc[0][1]);
    acc[1][0] = wmma_bf16(a1, b0, acc[1][0]);
    acc[1][1] = wmma_bf16(a1, b1, acc[1][1]);
  }
  int sel = lane >> 4, nn = lane & 15;
#pragma unroll
  for (int mt = 0; mt < 2; ++mt)
#pragma unroll
    for (int nt = 0; nt < 2; ++nt)
#pragma unroll
      for (int r = 0; r < 8; ++r) {
        long mg = m0 + mt * 16 + r + sel * 8;
        long ng = n0 + nt * 16 + nn;
        float v = acc[mt][nt][r] + bias[ng];
        long b = mg & (BS - 1), l = mg >> 1, h = ng >> 6, d = ng & 63;
        if (!isK) {
          v *= INV_T;
          outQ[(((b * NUM_HEAD + h) * SEQ + l) << 6) + d] = (bf16_t)v;
        } else {
          outKt[((b * NUM_HEAD + h) * HID + d) * SEQ + l] = (bf16_t)v;
        }
      }
}

// ---------------- kernel 2: ||k||^2 per (b,h,l) from Kt (coalesced over l)
__global__ void ksq_kernel(const bf16_t* __restrict__ Kt, float* __restrict__ ksq) {
  long idx = (long)blockIdx.x * blockDim.x + threadIdx.x;  // bh*SEQ + l
  if (idx >= (long)BH * SEQ) return;
  long bh = idx >> 11, l = idx & (SEQ - 1);
  const bf16_t* p = Kt + bh * HID * SEQ + l;
  float s = 0.f;
#pragma unroll 8
  for (int d = 0; d < HID; ++d) { float v = (float)p[(long)d * SEQ]; s += v * v; }
  ksq[idx] = s;
}

// ---------------- kernel 3: attention. block=128 (4 waves), wave = 32 queries.
// grid = (SEQ/128, BH). 3 passes over keys: max, sumexp, write attn + out accumulate.
// K (and V in pass 3) chunks are cooperatively staged in LDS via async b128 copies,
// double buffered, shared by the 4 waves.
__global__ void attn_kernel(const bf16_t* __restrict__ Qbf, const bf16_t* __restrict__ Kt,
                            const float* __restrict__ ksq, const bf16_t* __restrict__ comb,
                            float* __restrict__ attn_out, bf16_t* __restrict__ comb_out) {
  __shared__ __align__(16) bf16_t kbuf[2][HID * KP];     // [d][key]
  __shared__ __align__(16) bf16_t vbuf[2][KCH * KP];     // [key][d]
  __shared__ __align__(16) bf16_t lds_e[4][2][16 * 32];  // per-wave prob staging
  int tid  = threadIdx.x;
  int lane = tid & 31;
  int wid  = tid >> 5;
  long bh = blockIdx.y;            // b*16 + h
  long b  = bh >> 4, h = bh & 15;
  long q0 = (long)blockIdx.x * 128 + wid * 32;
  const bf16_t* Qh = Qbf + bh * SEQ * HID;
  const bf16_t* Kh = Kt + bh * HID * SEQ;
  const float*  kq = ksq + bh * SEQ;
  int sel = lane >> 4, nn = lane & 15;

  // Q fragments held across all passes: 2 m-tiles x 2 k-frags (d 0..31, 32..63)
  v16bf aq[2][2];
#pragma unroll
  for (int mt = 0; mt < 2; ++mt)
#pragma unroll
    for (int kf = 0; kf < 2; ++kf)
      aq[mt][kf] = load_a_bf16(Qh, HID, q0 + mt * 16, kf * 32, lane);

  // ---- pass 1: row max of scores s = 2*qk - |k|^2
  float pmax[2][8];
#pragma unroll
  for (int mt = 0; mt < 2; ++mt)
#pragma unroll
    for (int r = 0; r < 8; ++r) pmax[mt][r] = -1e30f;

  stage_k(Kh, 0, kbuf[0], tid);
  for (int ch = 0; ch < NCHUNK; ++ch) {
    const bf16_t* kb = kbuf[ch & 1];
    if (ch + 1 < NCHUNK) { stage_k(Kh, (ch + 1) * KCH, kbuf[(ch + 1) & 1], tid); WAIT_ASYNC4(); }
    else WAIT_ASYNC0();
    __syncthreads();
#pragma unroll
    for (int t16 = 0; t16 < 4; ++t16) {
      v16bf b0 = load_b_bf16(kb, KP, 0, t16 * 16, lane);
      v16bf b1 = load_b_bf16(kb, KP, 32, t16 * 16, lane);
      float kqv = kq[ch * KCH + t16 * 16 + nn];
#pragma unroll
      for (int mt = 0; mt < 2; ++mt) {
        v8f acc = {};
        acc = wmma_bf16(aq[mt][0], b0, acc);
        acc = wmma_bf16(aq[mt][1], b1, acc);
#pragma unroll
        for (int r = 0; r < 8; ++r)
          pmax[mt][r] = fmaxf(pmax[mt][r], 2.f * acc[r] - kqv);
      }
    }
    __syncthreads();
  }
#pragma unroll
  for (int mt = 0; mt < 2; ++mt)
#pragma unroll
    for (int r = 0; r < 8; ++r) {
      float v = pmax[mt][r];
      for (int off = 1; off < 16; off <<= 1) v = fmaxf(v, __shfl_xor(v, off, 16));
      pmax[mt][r] = v;
    }

  // ---- pass 2: row sum of exp(s - max)
  float psum[2][8] = {};
  stage_k(Kh, 0, kbuf[0], tid);
  for (int ch = 0; ch < NCHUNK; ++ch) {
    const bf16_t* kb = kbuf[ch & 1];
    if (ch + 1 < NCHUNK) { stage_k(Kh, (ch + 1) * KCH, kbuf[(ch + 1) & 1], tid); WAIT_ASYNC4(); }
    else WAIT_ASYNC0();
    __syncthreads();
#pragma unroll
    for (int t16 = 0; t16 < 4; ++t16) {
      v16bf b0 = load_b_bf16(kb, KP, 0, t16 * 16, lane);
      v16bf b1 = load_b_bf16(kb, KP, 32, t16 * 16, lane);
      float kqv = kq[ch * KCH + t16 * 16 + nn];
#pragma unroll
      for (int mt = 0; mt < 2; ++mt) {
        v8f acc = {};
        acc = wmma_bf16(aq[mt][0], b0, acc);
        acc = wmma_bf16(aq[mt][1], b1, acc);
#pragma unroll
        for (int r = 0; r < 8; ++r)
          psum[mt][r] += __expf(2.f * acc[r] - kqv - pmax[mt][r]);
      }
    }
    __syncthreads();
  }
  float rinv[2][8];
#pragma unroll
  for (int mt = 0; mt < 2; ++mt)
#pragma unroll
    for (int r = 0; r < 8; ++r) {
      float v = psum[mt][r];
      for (int off = 1; off < 16; off <<= 1) v += __shfl_xor(v, off, 16);
      rinv[mt][r] = 1.f / v;
    }

  // ---- pass 3: write normalized attn (f32) and accumulate out = attn @ V
  v8f oacc[2][4] = {};  // m-tile x d-tile(16)
  float* attn_base = attn_out + bh * SEQ * SEQ;
  stage_k(Kh, 0, kbuf[0], tid);
  stage_v(comb, b, h, 0, vbuf[0], tid);
  for (int ch = 0; ch < NCHUNK; ++ch) {
    const bf16_t* kb = kbuf[ch & 1];
    const bf16_t* vb = vbuf[ch & 1];
    if (ch + 1 < NCHUNK) {
      stage_k(Kh, (ch + 1) * KCH, kbuf[(ch + 1) & 1], tid);
      stage_v(comb, b, h, (ch + 1) * KCH, vbuf[(ch + 1) & 1], tid);
      WAIT_ASYNC8();
    } else WAIT_ASYNC0();
    __syncthreads();
#pragma unroll
    for (int sub = 0; sub < 2; ++sub) {      // 32-key halves of the chunk
#pragma unroll
      for (int kk = 0; kk < 2; ++kk) {       // 16-key score tiles
        int t16 = sub * 2 + kk;
        v16bf b0 = load_b_bf16(kb, KP, 0, t16 * 16, lane);
        v16bf b1 = load_b_bf16(kb, KP, 32, t16 * 16, lane);
        float kqv = kq[ch * KCH + t16 * 16 + nn];
#pragma unroll
        for (int mt = 0; mt < 2; ++mt) {
          v8f acc = {};
          acc = wmma_bf16(aq[mt][0], b0, acc);
          acc = wmma_bf16(aq[mt][1], b1, acc);
#pragma unroll
          for (int r = 0; r < 8; ++r) {
            float p = __expf(2.f * acc[r] - kqv - pmax[mt][r]) * rinv[mt][r];
            int m = r + sel * 8;
            attn_base[(q0 + mt * 16 + m) * SEQ + ch * KCH + t16 * 16 + nn] = p;
            lds_e[wid][mt][m * 32 + kk * 16 + nn] = (bf16_t)p;  // C->A relayout
          }
        }
      }
      asm volatile("s_wait_dscnt 0" ::: "memory");  // own-wave LDS RAW on lds_e
      v16bf ap0 = load_a_bf16(&lds_e[wid][0][0], 32, 0, 0, lane);
      v16bf ap1 = load_a_bf16(&lds_e[wid][1][0], 32, 0, 0, lane);
#pragma unroll
      for (int nt = 0; nt < 4; ++nt) {
        const bf16_t* p = vb + (sub * 32 + ((lane >> 4) << 3)) * KP + nt * 16 + (lane & 15);
        v16bf bv;
#pragma unroll
        for (int j = 0; j < 16; ++j) bv[j] = p[(j + (j & 8)) * KP];
        oacc[0][nt] = wmma_bf16(ap0, bv, oacc[0][nt]);
        oacc[1][nt] = wmma_bf16(ap1, bv, oacc[1][nt]);
      }
    }
    __syncthreads();
  }
  // store out (already normalized) as bf16 into combined's left half
#pragma unroll
  for (int mt = 0; mt < 2; ++mt)
#pragma unroll
    for (int nt = 0; nt < 4; ++nt)
#pragma unroll
      for (int r = 0; r < 8; ++r) {
        long qg = q0 + mt * 16 + r + sel * 8;
        comb_out[(qg * BS + b) * (2 * D_MODEL) + h * HID + nt * 16 + nn] =
            (bf16_t)oacc[mt][nt][r];
      }
}

// ---------------- kernel 4: final projection (M=4096,N=1024,K=2048), f32 out
__global__ void proj_out_kernel(const bf16_t* __restrict__ A, const bf16_t* __restrict__ Wt,
                                const float* __restrict__ bias, float* __restrict__ out) {
  int lane = threadIdx.x & 31;
  int wid  = threadIdx.x >> 5;
  long m0 = (long)blockIdx.x * 64 + (wid >> 1) * 32;
  long n0 = (long)blockIdx.y * 64 + (wid & 1) * 32;
  v8f acc[2][2] = {};
  for (int k = 0; k < 2 * D_MODEL; k += 32) {
    v16bf a0 = load_a_bf16(A, 2 * D_MODEL, m0, k, lane);
    v16bf a1 = load_a_bf16(A, 2 * D_MODEL, m0 + 16, k, lane);
    v16bf b0 = load_b_bf16(Wt, D_MODEL, k, n0, lane);
    v16bf b1 = load_b_bf16(Wt, D_MODEL, k, n0 + 16, lane);
    acc[0][0] = wmma_bf16(a0, b0, acc[0][0]);
    acc[0][1] = wmma_bf16(a0, b1, acc[0][1]);
    acc[1][0] = wmma_bf16(a1, b0, acc[1][0]);
    acc[1][1] = wmma_bf16(a1, b1, acc[1][1]);
  }
  int sel = lane >> 4, nn = lane & 15;
#pragma unroll
  for (int mt = 0; mt < 2; ++mt)
#pragma unroll
    for (int nt = 0; nt < 2; ++nt)
#pragma unroll
      for (int r = 0; r < 8; ++r) {
        long mg = m0 + mt * 16 + r + sel * 8;
        long ng = n0 + nt * 16 + nn;
        out[mg * D_MODEL + ng] = acc[mt][nt][r] + bias[ng];
      }
}

extern "C" void kernel_launch(void* const* d_in, const int* in_sizes, int n_in,
                              void* d_out, int out_size, void* d_ws, size_t ws_size,
                              hipStream_t stream) {
  (void)in_sizes; (void)n_in; (void)out_size; (void)ws_size;
  const float* Q  = (const float*)d_in[0];
  const float* K  = (const float*)d_in[1];
  const float* V  = (const float*)d_in[2];
  const float* WQ = (const float*)d_in[3];
  const float* bQ = (const float*)d_in[4];
  const float* WK = (const float*)d_in[5];
  const float* bK = (const float*)d_in[6];
  const float* Wp = (const float*)d_in[7];
  const float* bp = (const float*)d_in[8];
  float* out = (float*)d_out;

  char* ws = (char*)d_ws;
  bf16_t* WQt  = (bf16_t*)(ws + 0);                // 2 MB
  bf16_t* WKt  = (bf16_t*)(ws + (2ull << 20));     // 2 MB
  bf16_t* Wpt  = (bf16_t*)(ws + (4ull << 20));     // 4 MB
  bf16_t* Qbf  = (bf16_t*)(ws + (8ull << 20));     // 8 MB  [b][h][l][d]
  bf16_t* Kt   = (bf16_t*)(ws + (16ull << 20));    // 8 MB  [b][h][d][l]
  bf16_t* comb = (bf16_t*)(ws + (24ull << 20));    // 16 MB [4096][2048]
  float*  ksq  = (float*) (ws + (40ull << 20));    // 256 KB

  wt_bf16_kernel<<<4096, 256, 0, stream>>>(WQ, WQt, 1024, 1024);
  wt_bf16_kernel<<<4096, 256, 0, stream>>>(WK, WKt, 1024, 1024);
  wt_bf16_kernel<<<8192, 256, 0, stream>>>(Wp, Wpt, 1024, 2048);
  vcast_kernel<<<16384, 256, 0, stream>>>(V, comb);
  proj_qk_kernel<<<dim3(64, 16), 128, 0, stream>>>(Q, WQt, bQ, Qbf, Kt, 0);
  proj_qk_kernel<<<dim3(64, 16), 128, 0, stream>>>(K, WKt, bK, Qbf, Kt, 1);
  ksq_kernel<<<256, 256, 0, stream>>>(Kt, ksq);
  attn_kernel<<<dim3(SEQ / 128, BH), 128, 0, stream>>>(
      Qbf, Kt, ksq, comb, out + (size_t)MROWS * D_MODEL, comb);
  proj_out_kernel<<<dim3(64, 16), 128, 0, stream>>>(comb, Wpt, bp, out);
}
